// MultiHeadAttention_76647986365040
// MI455X (gfx1250) — matrix operand order repaired and checked by reading, not compiled, not run on hardware
//
#include <hip/hip_runtime.h>
#include <hip/hip_bf16.h>

#define B_  2
#define S_  2048
#define D_  1024
#define H_  16
#define HD_ 64

typedef __attribute__((ext_vector_type(16))) __bf16 bf16x16;
typedef __attribute__((ext_vector_type(8)))  float fx8;

union FragAB { bf16x16 v; unsigned int u[8]; };
union FragC  { fx8 v; float f[8]; };
union Pack8  { uint4 u4; unsigned short us[8]; };

__device__ __forceinline__ unsigned short f2bf_bits(float f) {
  unsigned int u = __float_as_uint(f);
  u += 0x7FFFu + ((u >> 16) & 1u);   // round to nearest even
  return (unsigned short)(u >> 16);
}

__device__ __forceinline__ unsigned int ldpair(const unsigned short* p) {
  return *reinterpret_cast<const unsigned int*>(p);
}

__device__ __forceinline__ fx8 wmma_bf16(bf16x16 a, bf16x16 b, fx8 c) {
  return __builtin_amdgcn_wmma_f32_16x16x32_bf16(false, a, false, b, (short)0, c, false, false);
}

// CDNA5 async DMA: global -> LDS, tracked by ASYNCcnt (no VGPR transit).
__device__ __forceinline__ void async_g2l_b128(unsigned int ldsoff, const void* gaddr) {
  asm volatile("global_load_async_to_lds_b128 %0, %1, off"
               :: "v"(ldsoff), "v"((unsigned long long)(size_t)gaddr) : "memory");
}
__device__ __forceinline__ void wait_async0() {
  asm volatile("s_wait_asynccnt 0" ::: "memory");
}
__device__ __forceinline__ unsigned int lds_off(const void* p) {
  return (unsigned int)(size_t)p;
}

// ---------------------------------------------------------------------------
// Prep kernel A: flat f32 -> bf16 (8 elements / thread).
// ---------------------------------------------------------------------------
__global__ __launch_bounds__(256)
void cvt_bf16_kernel(const float* __restrict__ src, unsigned short* __restrict__ dst) {
  size_t i = ((size_t)blockIdx.x * 256 + threadIdx.x) * 8;
  float4 a = *reinterpret_cast<const float4*>(&src[i]);
  float4 b = *reinterpret_cast<const float4*>(&src[i + 4]);
  Pack8 p;
  p.us[0] = f2bf_bits(a.x); p.us[1] = f2bf_bits(a.y);
  p.us[2] = f2bf_bits(a.z); p.us[3] = f2bf_bits(a.w);
  p.us[4] = f2bf_bits(b.x); p.us[5] = f2bf_bits(b.y);
  p.us[6] = f2bf_bits(b.z); p.us[7] = f2bf_bits(b.w);
  *reinterpret_cast<uint4*>(&dst[i]) = p.u4;
}

// ---------------------------------------------------------------------------
// Prep kernel B: weight transpose+convert, W[k][n] f32 -> Wt[n][k] bf16.
// 64x64 tile via LDS; z selects which of the 4 weights.
// ---------------------------------------------------------------------------
__global__ __launch_bounds__(256)
void wt_transpose_kernel(const float* __restrict__ Wq, const float* __restrict__ Wk,
                         const float* __restrict__ Wv, const float* __restrict__ Wo,
                         unsigned short* __restrict__ wqt, unsigned short* __restrict__ wkt,
                         unsigned short* __restrict__ wvt, unsigned short* __restrict__ wot) {
  const int z = blockIdx.z;
  const float* W = (z == 0) ? Wq : (z == 1) ? Wk : (z == 2) ? Wv : Wo;
  unsigned short* Wt = (z == 0) ? wqt : (z == 1) ? wkt : (z == 2) ? wvt : wot;

  const int n0 = blockIdx.x * 64;
  const int k0 = blockIdx.y * 64;
  const int tid = threadIdx.x;

  __shared__ unsigned short T[64][72];      // T[n][k]

#pragma unroll
  for (int i = 0; i < 4; ++i) {             // read 64x64 f32 coalesced, scatter to LDS
    int idx = tid + i * 256;                // 0..1023
    int kr = idx >> 4, c4 = (idx & 15) << 2;
    float4 f = *reinterpret_cast<const float4*>(&W[(size_t)(k0 + kr) * D_ + n0 + c4]);
    T[c4 + 0][kr] = f2bf_bits(f.x);
    T[c4 + 1][kr] = f2bf_bits(f.y);
    T[c4 + 2][kr] = f2bf_bits(f.z);
    T[c4 + 3][kr] = f2bf_bits(f.w);
  }
  __syncthreads();
#pragma unroll
  for (int i = 0; i < 2; ++i) {             // write transposed, coalesced bf16 rows
    int idx = tid + i * 256;                // 0..511
    int n = idx >> 3, k8 = (idx & 7) << 3;
    uint4 u = *reinterpret_cast<const uint4*>(&T[n][k8]);
    *reinterpret_cast<uint4*>(&Wt[(size_t)(n0 + n) * D_ + k0 + k8]) = u;
  }
}

// ---------------------------------------------------------------------------
// Kernel 1: fused QKV projections. 128x128 tile / WG, 64x64 per wave.
// A and B tiles are pure async DMA (bf16, pre-converted / pre-transposed).
// Q,K written [B,H,S,HD]; V written head-transposed [B,H,HD,S].
// ---------------------------------------------------------------------------
__global__ __launch_bounds__(128)
void qkv_proj_kernel(const unsigned short* __restrict__ xbf,
                     const unsigned short* __restrict__ wqt, const float* __restrict__ bq,
                     const unsigned short* __restrict__ wkt, const float* __restrict__ bk,
                     const unsigned short* __restrict__ wvt, const float* __restrict__ bv,
                     unsigned short* __restrict__ qh, unsigned short* __restrict__ kh,
                     unsigned short* __restrict__ vt) {
  const int z = blockIdx.z;
  const unsigned short* X  = xbf + (size_t)z * ((size_t)B_ * S_ * D_);
  const unsigned short* Wt = (z == 0) ? wqt : (z == 1) ? wkt : wvt;
  const float* bias        = (z == 0) ? bq  : (z == 1) ? bk  : bv;

  const int m0 = blockIdx.x * 128;
  const int n0 = blockIdx.y * 128;
  const int tid = threadIdx.x;
  const int wave = tid >> 5, lane = tid & 31;
  const int hgrp = lane >> 4, lm = lane & 15;
  const int wm = wave >> 1, wn = wave & 1;

  __shared__ unsigned short Al[2][128][40];   // [m][k]
  __shared__ unsigned short Bl[2][128][40];   // [n][k]

  FragC acc[4][4];
#pragma unroll
  for (int mt = 0; mt < 4; ++mt)
#pragma unroll
    for (int nt = 0; nt < 4; ++nt)
#pragma unroll
      for (int r = 0; r < 8; ++r) acc[mt][nt].f[r] = 0.0f;

  auto issue_tiles = [&](int kk, int b) {
#pragma unroll
    for (int i = 0; i < 4; ++i) {             // A: 128 rows x 32 k
      int idx = tid + i * 128;                // 0..511
      int row = idx >> 2, k8 = (idx & 3) << 3;
      async_g2l_b128(lds_off(&Al[b][row][k8]),
                     &X[((size_t)(m0 + row) << 10) + kk + k8]);
    }
#pragma unroll
    for (int i = 0; i < 4; ++i) {             // B: 128 n-rows x 32 k
      int idx = tid + i * 128;
      int row = idx >> 2, k8 = (idx & 3) << 3;
      async_g2l_b128(lds_off(&Bl[b][row][k8]),
                     &Wt[((size_t)(n0 + row) << 10) + kk + k8]);
    }
  };

  issue_tiles(0, 0);
  wait_async0();
  __syncthreads();

  int buf = 0;
  for (int kk = 32; kk <= D_; kk += 32) {
    if (kk < D_) issue_tiles(kk, buf ^ 1);    // DMA next tile while computing

    FragAB af[4];
#pragma unroll
    for (int mt = 0; mt < 4; ++mt)
#pragma unroll
      for (int vv = 0; vv < 8; ++vv) {
        int kof = ((vv >> 2) << 4) + (hgrp << 3) + ((vv & 3) << 1);
        af[mt].u[vv] = ldpair(&Al[buf][wm * 64 + mt * 16 + lm][kof]);
      }
#pragma unroll
    for (int nt = 0; nt < 4; ++nt) {
      FragAB bf_;
#pragma unroll
      for (int vv = 0; vv < 8; ++vv)
        bf_.u[vv] = ldpair(&Bl[buf][wn * 64 + nt * 16 + lm][((vv >> 2) << 4) + (hgrp << 3) + ((vv & 3) << 1)]);
#pragma unroll
      for (int mt = 0; mt < 4; ++mt)
        acc[mt][nt].v = wmma_bf16(af[mt].v, bf_.v, acc[mt][nt].v);
    }

    wait_async0();
    __syncthreads();
    buf ^= 1;
  }

#pragma unroll
  for (int nt = 0; nt < 4; ++nt) {
    int col = n0 + wn * 64 + nt * 16 + lm;
    float bb = bias[col];
    int hh = col >> 6, hd = col & 63;
#pragma unroll
    for (int mt = 0; mt < 4; ++mt)
#pragma unroll
      for (int r = 0; r < 8; ++r) {
        int row = m0 + wm * 64 + mt * 16 + r + hgrp * 8;
        int b = row >> 11;
        int s = row & (S_ - 1);
        unsigned short val = f2bf_bits(acc[mt][nt].f[r] + bb);
        if (z == 2)   // V: [B,H,HD,S] so attention can DMA d-major tiles
          vt[(((size_t)(b * H_ + hh) << 6) + hd) * S_ + s] = val;
        else if (z == 0)
          qh[(((size_t)(b * H_ + hh) * S_ + s) << 6) + hd] = val;
        else
          kh[(((size_t)(b * H_ + hh) * S_ + s) << 6) + hd] = val;
      }
  }
}

// ---------------------------------------------------------------------------
// Kernel 2: flash attention. One block = (b, h, 64 queries); 4 waves x 16 q.
// Q, K, V tiles are all async DMA (V is pre-transposed), double buffered.
// ---------------------------------------------------------------------------
__global__ __launch_bounds__(128)
void attn_kernel(const unsigned short* __restrict__ qh,
                 const unsigned short* __restrict__ kh,
                 const unsigned short* __restrict__ vt,
                 const int* __restrict__ mask,
                 unsigned short* __restrict__ xa) {
  const int qb = blockIdx.x;
  const int h  = blockIdx.y;
  const int bz = blockIdx.z;
  const int tid = threadIdx.x;
  const int wave = tid >> 5, lane = tid & 31;
  const int hgrp = lane >> 4, lm = lane & 15;

  __shared__ unsigned short Ql[64][72];       // [q][d]
  __shared__ unsigned short Kl[2][64][72];    // [key][d]
  __shared__ unsigned short Vl[2][64][72];    // [d][key]
  __shared__ unsigned short Pl[4][16][72];    // per-wave P, [q][key]

  const size_t headbase  = ((size_t)(bz * H_ + h) * S_) << 6;   // for qh/kh
  const size_t headbaseV = ((size_t)(bz * H_ + h) << 6) * S_;   // for vt

#pragma unroll
  for (int i = 0; i < 4; ++i) {               // async stage Q block 64x64
    int idx = tid + i * 128;
    int row = idx >> 3, d8 = (idx & 7) << 3;
    async_g2l_b128(lds_off(&Ql[row][d8]),
                   &qh[headbase + ((size_t)(qb * 64 + row) << 6) + d8]);
  }

  auto issueKV = [&](int kb, int b) {
#pragma unroll
    for (int i = 0; i < 4; ++i) {             // K: [key][d]
      int idx = tid + i * 128;
      int row = idx >> 3, d8 = (idx & 7) << 3;
      async_g2l_b128(lds_off(&Kl[b][row][d8]),
                     &kh[headbase + ((size_t)(kb * 64 + row) << 6) + d8]);
    }
#pragma unroll
    for (int i = 0; i < 4; ++i) {             // V: [d][key] straight from [B,H,HD,S]
      int idx = tid + i * 128;
      int d = idx >> 3, key8 = (idx & 7) << 3;
      async_g2l_b128(lds_off(&Vl[b][d][key8]),
                     &vt[headbaseV + (size_t)d * S_ + kb * 64 + key8]);
    }
  };

  issueKV(0, 0);
  wait_async0();
  __syncthreads();

  FragAB qf[2];                               // Q A-fragments (d=0..31, 32..63)
#pragma unroll
  for (int ks = 0; ks < 2; ++ks)
#pragma unroll
    for (int vv = 0; vv < 8; ++vv) {
      int kof = ks * 32 + ((vv >> 2) << 4) + (hgrp << 3) + ((vv & 3) << 1);
      qf[ks].u[vv] = ldpair(&Ql[wave * 16 + lm][kof]);
    }

  float m_state[8], l_state[8];
  FragC o[4];
#pragma unroll
  for (int r = 0; r < 8; ++r) { m_state[r] = -1e30f; l_state[r] = 0.0f; }
#pragma unroll
  for (int t = 0; t < 4; ++t)
#pragma unroll
    for (int r = 0; r < 8; ++r) o[t].f[r] = 0.0f;

  const float sm_scale = 0.125f;
  const int q_wave_row0 = qb * 64 + wave * 16;

  int buf = 0;
  for (int kb = 0; kb < S_ / 64; ++kb) {
    if (kb + 1 < S_ / 64) issueKV(kb + 1, buf ^ 1);

    FragC sc[4];
#pragma unroll
    for (int t = 0; t < 4; ++t) {
#pragma unroll
      for (int r = 0; r < 8; ++r) sc[t].f[r] = 0.0f;
#pragma unroll
      for (int ks = 0; ks < 2; ++ks) {
        FragAB kf;
#pragma unroll
        for (int vv = 0; vv < 8; ++vv) {
          int kof = ks * 32 + ((vv >> 2) << 4) + (hgrp << 3) + ((vv & 3) << 1);
          kf.u[vv] = ldpair(&Kl[buf][t * 16 + lm][kof]);
        }
        sc[t].v = wmma_bf16(qf[ks].v, kf.v, sc[t].v);
      }
    }

    const int kcol0 = kb * 64;
#pragma unroll
    for (int t = 0; t < 4; ++t) {             // scale + mask
      int kcol = kcol0 + t * 16 + lm;
#pragma unroll
      for (int r = 0; r < 8; ++r) {
        int qrow = q_wave_row0 + r + hgrp * 8;
        int mv = mask[(size_t)bz * S_ * S_ + (size_t)qrow * S_ + kcol];
        sc[t].f[r] = sc[t].f[r] * sm_scale + ((mv == 0) ? -1e9f : 0.0f);
      }
    }

#pragma unroll
    for (int r = 0; r < 8; ++r) {             // online softmax
      float mx = fmaxf(fmaxf(sc[0].f[r], sc[1].f[r]), fmaxf(sc[2].f[r], sc[3].f[r]));
#pragma unroll
      for (int mm = 1; mm < 16; mm <<= 1) mx = fmaxf(mx, __shfl_xor(mx, mm, 32));
      float mnew  = fmaxf(m_state[r], mx);
      float alpha = __expf(m_state[r] - mnew);
      float rsum = 0.0f;
#pragma unroll
      for (int t = 0; t < 4; ++t) {
        float p = __expf(sc[t].f[r] - mnew);
        sc[t].f[r] = p;
        rsum += p;
      }
#pragma unroll
      for (int mm = 1; mm < 16; mm <<= 1) rsum += __shfl_xor(rsum, mm, 32);
      l_state[r] = l_state[r] * alpha + rsum;
      m_state[r] = mnew;
#pragma unroll
      for (int t = 0; t < 4; ++t) o[t].f[r] *= alpha;
    }

#pragma unroll
    for (int t = 0; t < 4; ++t)               // C-layout -> A-layout via LDS
#pragma unroll
      for (int r = 0; r < 8; ++r)
        Pl[wave][r + hgrp * 8][t * 16 + lm] = f2bf_bits(sc[t].f[r]);

#pragma unroll
    for (int ks = 0; ks < 2; ++ks) {          // O += P . Vblk
      FragAB pf;
#pragma unroll
      for (int vv = 0; vv < 8; ++vv) {
        int kof = ks * 32 + ((vv >> 2) << 4) + (hgrp << 3) + ((vv & 3) << 1);
        pf.u[vv] = ldpair(&Pl[wave][lm][kof]);
      }
#pragma unroll
      for (int t = 0; t < 4; ++t) {
        FragAB vf;
#pragma unroll
        for (int vv = 0; vv < 8; ++vv) {
          int kof = ks * 32 + ((vv >> 2) << 4) + (hgrp << 3) + ((vv & 3) << 1);
          vf.u[vv] = ldpair(&Vl[buf][t * 16 + lm][kof]);
        }
        o[t].v = wmma_bf16(pf.v, vf.v, o[t].v);
      }
    }

    wait_async0();
    __syncthreads();
    buf ^= 1;
  }

#pragma unroll
  for (int t = 0; t < 4; ++t) {               // write attn output [B,S,D] bf16
    int d = t * 16 + lm;
#pragma unroll
    for (int r = 0; r < 8; ++r) {
      int qrow = q_wave_row0 + r + hgrp * 8;
      float val = o[t].f[r] / l_state[r];
      xa[((size_t)(bz * S_ + qrow) << 10) + h * 64 + d] = f2bf_bits(val);
    }
  }
}

// ---------------------------------------------------------------------------
// Kernel 3: output projection. All-async tiles (Xa bf16, Wo pre-transposed).
// ---------------------------------------------------------------------------
__global__ __launch_bounds__(128)
void out_proj_kernel(const unsigned short* __restrict__ xa,
                     const unsigned short* __restrict__ wot,
                     const float* __restrict__ bo,
                     float* __restrict__ out) {
  const int m0 = blockIdx.x * 128;
  const int n0 = blockIdx.y * 128;
  const int tid = threadIdx.x;
  const int wave = tid >> 5, lane = tid & 31;
  const int hgrp = lane >> 4, lm = lane & 15;
  const int wm = wave >> 1, wn = wave & 1;

  __shared__ unsigned short Al[2][128][40];
  __shared__ unsigned short Bl[2][128][40];

  FragC acc[4][4];
#pragma unroll
  for (int mt = 0; mt < 4; ++mt)
#pragma unroll
    for (int nt = 0; nt < 4; ++nt)
#pragma unroll
      for (int r = 0; r < 8; ++r) acc[mt][nt].f[r] = 0.0f;

  auto issue_tiles = [&](int kk, int b) {
#pragma unroll
    for (int i = 0; i < 4; ++i) {
      int idx = tid + i * 128;
      int row = idx >> 2, k8 = (idx & 3) << 3;
      async_g2l_b128(lds_off(&Al[b][row][k8]),
                     &xa[((size_t)(m0 + row) << 10) + kk + k8]);
    }
#pragma unroll
    for (int i = 0; i < 4; ++i) {
      int idx = tid + i * 128;
      int row = idx >> 2, k8 = (idx & 3) << 3;
      async_g2l_b128(lds_off(&Bl[b][row][k8]),
                     &wot[((size_t)(n0 + row) << 10) + kk + k8]);
    }
  };

  issue_tiles(0, 0);
  wait_async0();
  __syncthreads();

  int buf = 0;
  for (int kk = 32; kk <= D_; kk += 32) {
    if (kk < D_) issue_tiles(kk, buf ^ 1);

    FragAB af[4];
#pragma unroll
    for (int mt = 0; mt < 4; ++mt)
#pragma unroll
      for (int vv = 0; vv < 8; ++vv) {
        int kof = ((vv >> 2) << 4) + (hgrp << 3) + ((vv & 3) << 1);
        af[mt].u[vv] = ldpair(&Al[buf][wm * 64 + mt * 16 + lm][kof]);
      }
#pragma unroll
    for (int nt = 0; nt < 4; ++nt) {
      FragAB bf_;
#pragma unroll
      for (int vv = 0; vv < 8; ++vv)
        bf_.u[vv] = ldpair(&Bl[buf][wn * 64 + nt * 16 + lm][((vv >> 2) << 4) + (hgrp << 3) + ((vv & 3) << 1)]);
#pragma unroll
      for (int mt = 0; mt < 4; ++mt)
        acc[mt][nt].v = wmma_bf16(af[mt].v, bf_.v, acc[mt][nt].v);
    }

    wait_async0();
    __syncthreads();
    buf ^= 1;
  }

#pragma unroll
  for (int nt = 0; nt < 4; ++nt) {
    int col = n0 + wn * 64 + nt * 16 + lm;
    float bb = bo[col];
#pragma unroll
    for (int mt = 0; mt < 4; ++mt)
#pragma unroll
      for (int r = 0; r < 8; ++r) {
        int row = m0 + wm * 64 + mt * 16 + r + hgrp * 8;
        out[((size_t)row << 10) + col] = acc[mt][nt].f[r] + bb;
      }
  }
}

// ---------------------------------------------------------------------------
extern "C" void kernel_launch(void* const* d_in, const int* in_sizes, int n_in,
                              void* d_out, int out_size, void* d_ws, size_t ws_size,
                              hipStream_t stream) {
  (void)in_sizes; (void)n_in; (void)out_size; (void)ws_size;
  const float* q  = (const float*)d_in[0];
  const float* k  = (const float*)d_in[1];
  const float* v  = (const float*)d_in[2];
  const int* mask = (const int*)d_in[3];
  const float* Wq = (const float*)d_in[4];
  const float* bq = (const float*)d_in[5];
  const float* Wk = (const float*)d_in[6];
  const float* bk = (const float*)d_in[7];
  const float* Wv = (const float*)d_in[8];
  const float* bv = (const float*)d_in[9];
  const float* Wo = (const float*)d_in[10];
  const float* bo = (const float*)d_in[11];
  float* out = (float*)d_out;

  const size_t NM  = (size_t)B_ * S_ * D_;         // 4M activations per tensor
  const size_t NW  = (size_t)D_ * D_;              // 1M per weight
  const size_t per = (size_t)B_ * H_ * S_ * HD_;   // 4M per head tensor

  unsigned short* xbf = (unsigned short*)d_ws;     // 3*NM
  unsigned short* wqt = xbf + 3 * NM;
  unsigned short* wkt = wqt + NW;
  unsigned short* wvt = wkt + NW;
  unsigned short* wot = wvt + NW;
  unsigned short* qh  = wot + NW;                  // [B,H,S,HD]
  unsigned short* kh  = qh + per;                  // [B,H,S,HD]
  unsigned short* vt  = kh + per;                  // [B,H,HD,S]
  unsigned short* xa  = vt + per;                  // [B,S,D]

  const int cvt_blocks = (int)(NM / (256 * 8));    // 2048
  cvt_bf16_kernel<<<cvt_blocks, 256, 0, stream>>>(q, xbf + 0 * NM);
  cvt_bf16_kernel<<<cvt_blocks, 256, 0, stream>>>(k, xbf + 1 * NM);
  cvt_bf16_kernel<<<cvt_blocks, 256, 0, stream>>>(v, xbf + 2 * NM);
  wt_transpose_kernel<<<dim3(16, 16, 4), 256, 0, stream>>>(Wq, Wk, Wv, Wo,
                                                           wqt, wkt, wvt, wot);

  dim3 blk(128);
  qkv_proj_kernel<<<dim3(32, 8, 3), blk, 0, stream>>>(xbf, wqt, bq, wkt, bk, wvt, bv,
                                                      qh, kh, vt);
  attn_kernel<<<dim3(S_ / 64, H_, B_), blk, 0, stream>>>(qh, kh, vt, mask, xa);
  out_proj_kernel<<<dim3(32, 8), blk, 0, stream>>>(xa, wot, bo, out);
}
